// SimilarityTreeLSTM_29154238005591
// MI455X (gfx1250) — matrix operand order repaired and checked by reading, not compiled, not run on hardware
//
#include <hip/hip_runtime.h>

// ---------------------------------------------------------------------------
// SimilarityTreeLSTM on MI455X (gfx1250, wave32, WMMA f32_16x16x32_f16).
// All GEMMs run on v_wmma_f32_16x16x32_f16 with f16 A/B (LDS-staged A,
// fragment-major packed B in workspace) and f32 accumulators. Cell state c
// is kept in f32 across the recurrence; h is requantized to f16 per level.
//
// WMMA inner loops are branch-free (N padded so each wave owns a fixed,
// contiguous tile set; whole-GEMM uniform guards where padding would waste
// work). B fragments use unsigned 32-bit element indices so loads lower to
// the SADDR + 32-bit-voffset form, and all B loads of a kt step are issued
// before the WMMA chain for latency hiding.
// ---------------------------------------------------------------------------

typedef __attribute__((ext_vector_type(16))) _Float16 v16h;
typedef __attribute__((ext_vector_type(8)))  float    v8f;

#define BDIM 256   // 8 wave32s per workgroup

// Problem constants (from the reference).
#define BQ   2048       // sentence pairs
#define TRES 4096       // 2*BQ trees
#define LL   64         // leaves per tree
// Padded tile dims.
#define K_LEAF 320      // IN=300 -> 320
#define K_M    160      // M=150  -> 160
#define N_IOU  512      // 3M=450 -> 512 (32 tiles: branch-free, 4/wave)
#define KT_LEAF 10      // 320/32
#define KT_M     5      // 160/32
#define NT_IOU  32      // 512/16
#define NT_F    10      // 160/16 (waves 0..4, 2 tiles each)
#define KT_HID  10      // 320/32 (2M concat features)
#define NT_HID   4      // 64/16  (HID=50 -> 64; waves 0..3, 1 tile each)

__device__ __forceinline__ float sigf(float x){ return 1.0f/(1.0f + __expf(-x)); }

// Wave id as an explicit SGPR so wave-uniform branches become scalar.
__device__ __forceinline__ int wave_id(){
  return __builtin_amdgcn_readfirstlane((int)(threadIdx.x >> 5));
}

// A-fragment loader, 16x32 f16 tile from LDS row-major [16][lda].
// Layout per ISA 7.12.2 (wave32): lane m=lane&15, half=lane>>4;
// elements 0..7 -> K = half*8 + e ; elements 8..15 -> K = 16 + half*8 + (e-8).
__device__ __forceinline__ v16h load_a_lds(const _Float16* As, int lda, int kt, int lane){
  const int m = lane & 15, half = lane >> 4;
  const _Float16* p = As + m*lda + kt*32 + half*8;
  v16h r;
  #pragma unroll
  for (int e = 0; e < 8; ++e) r[e] = p[e];
  #pragma unroll
  for (int e = 0; e < 8; ++e) r[8+e] = p[16+e];
  return r;
}

// B-fragment loader: weights pre-packed fragment-major, one contiguous
// 32-byte load per lane. Unsigned 32-bit element index -> SADDR+voffset form.
__device__ __forceinline__ v16h load_b_pk(const _Float16* __restrict__ Wpk,
                                          int kt, int nt, int NT, int lane){
  unsigned idx = ((unsigned)(kt*NT + nt) * 32u + (unsigned)lane) * 16u;
  return *(const v16h*)(Wpk + idx);
}

// Branch-free tiled GEMM core: wave owns tiles [tile0, tile0+TPW).
// NT is the total packed tile count (address stride). No conditionals in
// the kt loop -> accumulators coalesce in place across iterations. All B
// loads of a kt step are issued before the WMMA chain.
template<int NT, int KT, int LDA, int TPW>
__device__ __forceinline__ void wmma_gemm(const _Float16* As,
                                          const _Float16* __restrict__ wpk,
                                          int lane, int tile0, v8f (&acc)[TPW]){
  #pragma unroll
  for (int i = 0; i < TPW; ++i)
    #pragma unroll
    for (int e = 0; e < 8; ++e) acc[i][e] = 0.0f;
  for (int kt = 0; kt < KT; ++kt){
    v16h a = load_a_lds(As, LDA, kt, lane);
    v16h bb[TPW];
    #pragma unroll
    for (int i = 0; i < TPW; ++i)
      bb[i] = load_b_pk(wpk, kt, tile0 + i, NT, lane);
    #pragma unroll
    for (int i = 0; i < TPW; ++i)
      acc[i] = __builtin_amdgcn_wmma_f32_16x16x32_f16(false, a, false, bb[i],
                                                      (short)0, acc[i], false, false);
  }
}

// ---------------------------------------------------------------------------
// Weight packing: f32 [K][N] row-major -> f16 fragment-major (zero padded).
// B-operand hardware layout (ISA 7.12.2): lane n=lane&15, K = (lane>>4)*8 + e (e<8), 16 + (lane>>4)*8 + (e-8) (e>=8).
// ---------------------------------------------------------------------------
__global__ __launch_bounds__(BDIM) void pack_b_kernel(const float* __restrict__ W,
                                                      int K, int N, int KT, int NT,
                                                      _Float16* __restrict__ dst){
  int idx = blockIdx.x * BDIM + threadIdx.x;
  int total = KT * NT * 512;
  if (idx >= total) return;
  int e    =  idx        & 15;
  int lane = (idx >> 4)  & 31;
  int tile =  idx >> 9;
  int nt = tile % NT, kt = tile / NT;
  int k = kt*32 + (lane >> 4)*8 + (e < 8 ? e : 8 + e);
  int n = nt*16 + (lane & 15);
  float v = (k < K && n < N) ? W[(size_t)k*N + n] : 0.0f;
  dst[idx] = (_Float16)v;
}

__global__ __launch_bounds__(BDIM) void prep_bias_kernel(const float* __restrict__ bioux,
                                                         const float* __restrict__ biouh,
                                                         const float* __restrict__ bfh,
                                                         const float* __restrict__ bhv,
                                                         float* __restrict__ bleaf,
                                                         float* __restrict__ bint,
                                                         float* __restrict__ bfp,
                                                         float* __restrict__ bhp){
  int i = blockIdx.x * BDIM + threadIdx.x;
  if (i < N_IOU){
    float a = (i < 450) ? bioux[i] : 0.0f;
    float b = (i < 450) ? biouh[i] : 0.0f;
    bleaf[i] = a + b;     // leaves: child h = 0, iouh contributes only bias
    bint[i]  = b;
  }
  if (i < K_M) bfp[i] = (i < 150) ? bfh[i] : 0.0f;
  if (i < 64)  bhp[i] = (i < 50)  ? bhv[i] : 0.0f;
}

// ---------------------------------------------------------------------------
// Leaf kernel: gather embeddings into LDS (f32->f16), GEMM vs packed Wioux,
// epilogue through LDS to split i/o/u, write h (f16, pad-160) and c (f32).
// Rows: 2*B*L = 262144 -> 16384 tiles of 16.
// ---------------------------------------------------------------------------
__global__ __launch_bounds__(BDIM) void leaf_kernel(const int* __restrict__ lt,
                                                    const int* __restrict__ rt,
                                                    const float* __restrict__ emb,
                                                    const _Float16* __restrict__ wpk,
                                                    const float* __restrict__ bleaf,
                                                    _Float16* __restrict__ hout,
                                                    float* __restrict__ cout){
  __shared__ _Float16 As[16 * K_LEAF];
  __shared__ float    Cs[16 * N_IOU];
  const int row0 = blockIdx.x * 16;
  const int tid  = threadIdx.x;

  // Stage A: 16 gathered embedding rows, zero-padded to 320 cols.
  for (int idx = tid; idx < 16 * K_LEAF; idx += BDIM){
    int r = idx / K_LEAF, k = idx - r * K_LEAF;
    int grow = row0 + r;
    int t = grow >> 6, j = grow & 63;             // tree, leaf
    int tok = (t < BQ) ? lt[t*LL + j] : rt[(t - BQ)*LL + j];
    float v = (k < 300) ? emb[(size_t)tok * 300 + k] : 0.0f;
    As[idx] = (_Float16)v;
  }
  __syncthreads();

  const int lane = tid & 31;
  const int wave = wave_id();
  const int tile0 = wave * 4;
  v8f acc[4];
  wmma_gemm<NT_IOU, KT_LEAF, K_LEAF, 4>(As, wpk, lane, tile0, acc);

  // Spill C tiles (+bias) to LDS for cross-column i/o/u access.
  {
    const int half = lane >> 4, n = lane & 15;
    #pragma unroll
    for (int i = 0; i < 4; ++i){
      int col = (tile0 + i)*16 + n;
      #pragma unroll
      for (int r = 0; r < 8; ++r)
        Cs[(r + 8*half) * N_IOU + col] = acc[i][r] + bleaf[col];
    }
  }
  __syncthreads();

  // Leaf cell: c = sig(i)*tanh(u); h = sig(o)*tanh(c).
  for (int idx = tid; idx < 16 * K_M; idx += BDIM){
    int r = idx / K_M, m = idx - r * K_M;
    int grow = row0 + r;
    if (m < 150){
      float iv = Cs[r * N_IOU + m];
      float ov = Cs[r * N_IOU + 150 + m];
      float uv = Cs[r * N_IOU + 300 + m];
      float c = sigf(iv) * tanhf(uv);
      float h = sigf(ov) * tanhf(c);
      cout[(size_t)grow * 150 + m] = c;
      hout[(size_t)grow * 160 + m] = (_Float16)h;
    } else {
      hout[(size_t)grow * 160 + m] = (_Float16)0.0f;  // K-pad for next GEMM
    }
  }
}

// ---------------------------------------------------------------------------
// Forget-gate kernel: F = h_child @ Wfh; fbuf = sigmoid(F + bfh) * c_child.
// Rows = 2B*n (children). Waves 0..4 own 2 contiguous tiles each; single
// uniform guard around GEMM+epilogue (acc dead on skip path -> no phis).
// ---------------------------------------------------------------------------
__global__ __launch_bounds__(BDIM) void fgate_kernel(const _Float16* __restrict__ hc,
                                                     const float* __restrict__ cc,
                                                     const _Float16* __restrict__ wpk,
                                                     const float* __restrict__ bfp,
                                                     float* __restrict__ fbuf){
  __shared__ _Float16 As[16 * K_M];
  const int row0 = blockIdx.x * 16;
  const int tid  = threadIdx.x;

  for (int idx = tid; idx < 16 * K_M; idx += BDIM)
    As[idx] = hc[(size_t)(row0 + idx / K_M) * 160 + (idx % K_M)];
  __syncthreads();

  const int lane = tid & 31;
  const int wave = wave_id();
  if (wave < 5){
    const int tile0 = wave * 2;
    v8f acc[2];
    wmma_gemm<NT_F, KT_M, K_M, 2>(As, wpk, lane, tile0, acc);

    const int half = lane >> 4, n = lane & 15;
    #pragma unroll
    for (int i = 0; i < 2; ++i){
      int col = (tile0 + i)*16 + n;
      if (col < 150){
        #pragma unroll
        for (int r = 0; r < 8; ++r){
          int grow = row0 + r + 8*half;
          float f = sigf(acc[i][r] + bfp[col]);
          fbuf[(size_t)grow * 150 + col] = f * cc[(size_t)grow * 150 + col];
        }
      }
    }
  }
}

// ---------------------------------------------------------------------------
// Internal-node kernel: child-sum in A stager, GEMM vs packed Wiouh,
// epilogue combines with fbuf (= sig(f)*c per child) to produce parent c, h.
// Rows = 2B*(n/2) parents.
// ---------------------------------------------------------------------------
__global__ __launch_bounds__(BDIM) void internal_kernel(const _Float16* __restrict__ hc,
                                                        const float* __restrict__ fbuf,
                                                        const _Float16* __restrict__ wpk,
                                                        const float* __restrict__ bint,
                                                        _Float16* __restrict__ hn,
                                                        float* __restrict__ cn,
                                                        int n /* children per tree */){
  __shared__ _Float16 As[16 * K_M];
  __shared__ float    Cs[16 * N_IOU];
  const int nh   = n >> 1;
  const int row0 = blockIdx.x * 16;
  const int tid  = threadIdx.x;

  // Stage A = h[child0] + h[child1] (child-sum), f16.
  for (int idx = tid; idx < 16 * K_M; idx += BDIM){
    int r = idx / K_M, k = idx - r * K_M;
    int p = row0 + r;
    int t = p / nh, j = p - t * nh;
    size_t c0 = (size_t)(t*n + 2*j) * 160 + k;
    As[idx] = (_Float16)((float)hc[c0] + (float)hc[c0 + 160]);
  }
  __syncthreads();

  const int lane = tid & 31;
  const int wave = wave_id();
  const int tile0 = wave * 4;
  v8f acc[4];
  wmma_gemm<NT_IOU, KT_M, K_M, 4>(As, wpk, lane, tile0, acc);

  {
    const int half = lane >> 4, nn = lane & 15;
    #pragma unroll
    for (int i = 0; i < 4; ++i){
      int col = (tile0 + i)*16 + nn;
      #pragma unroll
      for (int r = 0; r < 8; ++r)
        Cs[(r + 8*half) * N_IOU + col] = acc[i][r] + bint[col];
    }
  }
  __syncthreads();

  for (int idx = tid; idx < 16 * K_M; idx += BDIM){
    int r = idx / K_M, m = idx - r * K_M;
    int p = row0 + r;
    if (m < 150){
      int t = p / nh, j = p - t * nh;
      size_t f0 = (size_t)(t*n + 2*j) * 150 + m;
      float iv = Cs[r * N_IOU + m];
      float ov = Cs[r * N_IOU + 150 + m];
      float uv = Cs[r * N_IOU + 300 + m];
      float c = sigf(iv) * tanhf(uv) + fbuf[f0] + fbuf[f0 + 150];
      float h = sigf(ov) * tanhf(c);
      cn[(size_t)p * 150 + m] = c;
      hn[(size_t)p * 160 + m] = (_Float16)h;
    } else {
      hn[(size_t)p * 160 + m] = (_Float16)0.0f;
    }
  }
}

// ---------------------------------------------------------------------------
// Head GEMM: features [lh*rh, |lh-rh|] built in the A stager (pad 300->320),
// mid = sigmoid(vec @ Wh + bh). Rows = B = 2048 -> 128 tiles. NT = 4.
// ---------------------------------------------------------------------------
__global__ __launch_bounds__(BDIM) void final_mid_kernel(const _Float16* __restrict__ hroot,
                                                         const _Float16* __restrict__ wpk,
                                                         const float* __restrict__ bhp,
                                                         float* __restrict__ mid){
  __shared__ _Float16 As[16 * K_LEAF];
  const int row0 = blockIdx.x * 16;
  const int tid  = threadIdx.x;

  for (int idx = tid; idx < 16 * K_LEAF; idx += BDIM){
    int r = idx / K_LEAF, k = idx - r * K_LEAF;
    int b = row0 + r;                  // pair index
    float v = 0.0f;
    if (k < 150){
      float lh = (float)hroot[(size_t)b * 160 + k];
      float rh = (float)hroot[(size_t)(b + BQ) * 160 + k];
      v = lh * rh;
    } else if (k < 300){
      int m = k - 150;
      float lh = (float)hroot[(size_t)b * 160 + m];
      float rh = (float)hroot[(size_t)(b + BQ) * 160 + m];
      v = fabsf(lh - rh);
    }
    As[idx] = (_Float16)v;
  }
  __syncthreads();

  const int lane = tid & 31;
  const int wave = wave_id();
  if (wave < NT_HID){
    v8f acc[1];
    wmma_gemm<NT_HID, KT_HID, K_LEAF, 1>(As, wpk, lane, wave, acc);

    const int half = lane >> 4, nn = lane & 15;
    int col = wave*16 + nn;
    if (col < 50){
      #pragma unroll
      for (int r = 0; r < 8; ++r){
        int grow = row0 + r + 8*half;
        mid[(size_t)grow * 50 + col] = sigf(acc[0][r] + bhp[col]);
      }
    }
  }
}

// Tiny classifier: z = mid @ Wp + bp (50x5), log_softmax. One thread per row.
__global__ __launch_bounds__(BDIM) void classify_kernel(const float* __restrict__ mid,
                                                        const float* __restrict__ Wp,
                                                        const float* __restrict__ bp,
                                                        float* __restrict__ out){
  int b = blockIdx.x * BDIM + threadIdx.x;
  if (b >= BQ) return;
  float z[5];
  #pragma unroll
  for (int c = 0; c < 5; ++c) z[c] = bp[c];
  for (int j = 0; j < 50; ++j){
    float m = mid[(size_t)b * 50 + j];
    #pragma unroll
    for (int c = 0; c < 5; ++c) z[c] += m * Wp[j*5 + c];
  }
  float mx = z[0];
  #pragma unroll
  for (int c = 1; c < 5; ++c) mx = fmaxf(mx, z[c]);
  float s = 0.0f;
  #pragma unroll
  for (int c = 0; c < 5; ++c) s += expf(z[c] - mx);
  float lse = mx + logf(s);
  #pragma unroll
  for (int c = 0; c < 5; ++c) out[(size_t)b * 5 + c] = z[c] - lse;
}

// ---------------------------------------------------------------------------
extern "C" void kernel_launch(void* const* d_in, const int* in_sizes, int n_in,
                              void* d_out, int out_size, void* d_ws, size_t ws_size,
                              hipStream_t stream){
  (void)in_sizes; (void)n_in; (void)out_size; (void)ws_size;
  const int*   lt    = (const int*)  d_in[0];
  const int*   rt    = (const int*)  d_in[1];
  const float* emb   = (const float*)d_in[2];
  const float* Wioux = (const float*)d_in[3];
  const float* bioux = (const float*)d_in[4];
  const float* Wiouh = (const float*)d_in[5];
  const float* biouh = (const float*)d_in[6];
  // d_in[7] = Wfx, d_in[8] = bfx: unused by the reference's _encode.
  const float* Wfh   = (const float*)d_in[9];
  const float* bfh   = (const float*)d_in[10];
  const float* Wh    = (const float*)d_in[11];
  const float* bh    = (const float*)d_in[12];
  const float* Wp    = (const float*)d_in[13];
  const float* bp    = (const float*)d_in[14];
  float* out = (float*)d_out;

  char* ws = (char*)d_ws;
  size_t off = 0;
  auto alloc = [&](size_t bytes) -> void* {
    void* p = (void*)(ws + off);
    off = (off + bytes + 255) & ~(size_t)255;
    return p;
  };

  _Float16* wioux_pk = (_Float16*)alloc((size_t)KT_LEAF * NT_IOU * 512 * 2);
  _Float16* wiouh_pk = (_Float16*)alloc((size_t)KT_M    * NT_IOU * 512 * 2);
  _Float16* wfh_pk   = (_Float16*)alloc((size_t)KT_M    * NT_F   * 512 * 2);
  _Float16* wh_pk    = (_Float16*)alloc((size_t)KT_HID  * NT_HID * 512 * 2);
  float* bleaf = (float*)alloc(N_IOU * 4);
  float* bint  = (float*)alloc(N_IOU * 4);
  float* bfp   = (float*)alloc(K_M * 4);
  float* bhp   = (float*)alloc(64 * 4);
  _Float16* h0 = (_Float16*)alloc((size_t)TRES * 64 * 160 * 2);
  _Float16* h1 = (_Float16*)alloc((size_t)TRES * 32 * 160 * 2);
  float* c0    = (float*)alloc((size_t)TRES * 64 * 150 * 4);
  float* c1    = (float*)alloc((size_t)TRES * 32 * 150 * 4);
  float* fbuf  = (float*)alloc((size_t)TRES * 64 * 150 * 4);
  float* midb  = (float*)alloc((size_t)BQ * 50 * 4);

  // Weight/bias prep (deterministic, per launch).
  {
    int tot;
    tot = KT_LEAF*NT_IOU*512; pack_b_kernel<<<(tot+BDIM-1)/BDIM, BDIM, 0, stream>>>(Wioux, 300, 450, KT_LEAF, NT_IOU, wioux_pk);
    tot = KT_M   *NT_IOU*512; pack_b_kernel<<<(tot+BDIM-1)/BDIM, BDIM, 0, stream>>>(Wiouh, 150, 450, KT_M,    NT_IOU, wiouh_pk);
    tot = KT_M   *NT_F  *512; pack_b_kernel<<<(tot+BDIM-1)/BDIM, BDIM, 0, stream>>>(Wfh,   150, 150, KT_M,    NT_F,   wfh_pk);
    tot = KT_HID *NT_HID*512; pack_b_kernel<<<(tot+BDIM-1)/BDIM, BDIM, 0, stream>>>(Wh,    300,  50, KT_HID,  NT_HID, wh_pk);
    prep_bias_kernel<<<2, BDIM, 0, stream>>>(bioux, biouh, bfh, bh, bleaf, bint, bfp, bhp);
  }

  // Leaves: 2B*L rows.
  leaf_kernel<<<(TRES * LL) / 16, BDIM, 0, stream>>>(lt, rt, emb, wioux_pk, bleaf, h0, c0);

  // Internal levels, 64 -> 1 (6 serial level pairs).
  _Float16* hc = h0; _Float16* hn = h1;
  float* cc = c0;    float* cn = c1;
  for (int n = 64; n >= 2; n >>= 1){
    int child_rows = TRES * n;
    fgate_kernel   <<<child_rows / 16,       BDIM, 0, stream>>>(hc, cc, wfh_pk, bfp, fbuf);
    internal_kernel<<<(child_rows / 2) / 16, BDIM, 0, stream>>>(hc, fbuf, wiouh_pk, bint, hn, cn, n);
    _Float16* th = hc; hc = hn; hn = th;
    float*    tc = cc; cc = cn; cn = tc;
  }

  // Similarity head + classifier.
  final_mid_kernel<<<BQ / 16, BDIM, 0, stream>>>(hc, wh_pk, bhp, midb);
  classify_kernel <<<(BQ + BDIM - 1) / BDIM, BDIM, 0, stream>>>(midb, Wp, bp, out);
}
